// LinearAttentionLayer_86260123174779
// MI455X (gfx1250) — matrix operand order repaired and checked by reading, not compiled, not run on hardware
//
#include <hip/hip_runtime.h>
#include <hip/hip_bf16.h>

#define D_MODEL 1024
#define N_HEADS 16
#define HEAD_D  64
#define BATCH   4
#define SEQ_L   2048
#define M_ROWS  (BATCH * SEQ_L)   /* 8192 */
#define N_CHUNK (SEQ_L / 64)      /* 32 */
#define EPS_CAUSAL 1e-6f

typedef __attribute__((ext_vector_type(8)))  __bf16 v8bf;
typedef __attribute__((ext_vector_type(16))) __bf16 v16bf;
typedef __attribute__((ext_vector_type(8)))  float  v8f;

__device__ __forceinline__ unsigned short f2bf(float f) {
    unsigned int u = __builtin_bit_cast(unsigned int, f);
    u += 0x7FFFu + ((u >> 16) & 1u);          // RNE
    return (unsigned short)(u >> 16);
}
__device__ __forceinline__ float bf2f(unsigned short u) {
    unsigned int x = ((unsigned int)u) << 16;
    return __builtin_bit_cast(float, x);
}

// 16x32 bf16 A-fragment from a stride-64 LDS tile: lane m=l%16, half=l/16;
// e<8 -> K = kt*32 + half*8 + e ; e>=8 -> K = kt*32 + 16 + half*8 + (e-8)
__device__ __forceinline__ v16bf ldA64(const unsigned short* T, int row, int kt, int half) {
    const int base = row * 64 + kt * 32 + half * 8;
    v8bf r1 = *reinterpret_cast<const v8bf*>(T + base);
    v8bf r2 = *reinterpret_cast<const v8bf*>(T + base + 16);
    v16bf a;
    #pragma unroll
    for (int i = 0; i < 8; ++i) { a[i] = r1[i]; a[i + 8] = r2[i]; }
    return a;
}
// 32x16 bf16 B-fragment from a stride-64 LDS tile: lane n=l%16,
// e<8 -> K = kt*32 + half*8 + e ; e>=8 -> K = kt*32 + 16 + half*8 + (e-8)
__device__ __forceinline__ v16bf ldB64(const unsigned short* T, int row, int kt, int half) {
    const int base = row * 64 + kt * 32 + half * 8;
    v8bf r1 = *reinterpret_cast<const v8bf*>(T + base);
    v8bf r2 = *reinterpret_cast<const v8bf*>(T + base + 16);
    v16bf b;
    #pragma unroll
    for (int i = 0; i < 8; ++i) { b[i] = r1[i]; b[i + 8] = r2[i]; }
    return b;
}
// Same fragment builders against stride-1024 global bf16 matrices.
__device__ __forceinline__ v16bf ldA_g(const unsigned short* rowp, int kk, int half) {
    const int base = kk + half * 8;
    v8bf r1 = *reinterpret_cast<const v8bf*>(rowp + base);
    v8bf r2 = *reinterpret_cast<const v8bf*>(rowp + base + 16);
    v16bf a;
    #pragma unroll
    for (int i = 0; i < 8; ++i) { a[i] = r1[i]; a[i + 8] = r2[i]; }
    return a;
}
__device__ __forceinline__ v16bf ldB_g(const unsigned short* rowp, int kk, int half) {
    const int base = kk + half * 8;
    v8bf r1 = *reinterpret_cast<const v8bf*>(rowp + base);
    v8bf r2 = *reinterpret_cast<const v8bf*>(rowp + base + 16);
    v16bf b;
    #pragma unroll
    for (int i = 0; i < 8; ++i) { b[i] = r1[i]; b[i + 8] = r2[i]; }
    return b;
}

#define WMMA_BF16(A, B, C) \
    __builtin_amdgcn_wmma_f32_16x16x32_bf16(false, (A), false, (B), (short)0, (C), false, false)

// ---------------------------------------------------------------------------
// Prep 1: elementwise fp32 -> bf16
// ---------------------------------------------------------------------------
__global__ __launch_bounds__(256) void cvt_bf16_kernel(const float* __restrict__ X,
                                                       unsigned short* __restrict__ Y, int n) {
    int i = (blockIdx.x * 256 + threadIdx.x) * 4;
    if (i + 3 < n) {
        float4 x = *reinterpret_cast<const float4*>(&X[i]);
        unsigned int lo = (unsigned int)f2bf(x.x) | ((unsigned int)f2bf(x.y) << 16);
        unsigned int hi = (unsigned int)f2bf(x.z) | ((unsigned int)f2bf(x.w) << 16);
        uint2 p; p.x = lo; p.y = hi;
        *reinterpret_cast<uint2*>(&Y[i]) = p;
    }
}

// ---------------------------------------------------------------------------
// Prep 2: W[1024k][1024n] fp32 -> WT[1024n][1024k] bf16 (LDS tile transpose)
// ---------------------------------------------------------------------------
__global__ __launch_bounds__(256) void transpose_w_kernel(const float* __restrict__ W,
                                                          unsigned short* __restrict__ WT) {
    __shared__ unsigned short t[32][33];
    const int tid = threadIdx.x;
    const int kb = blockIdx.x * 32;          // K tile base
    const int nb = blockIdx.y * 32;          // N tile base
    const int r  = tid >> 3;                 // 0..31
    const int c4 = (tid & 7) * 4;            // 0,4,..,28
    float4 w = *reinterpret_cast<const float4*>(&W[(size_t)(kb + r) * 1024 + nb + c4]);
    t[r][c4 + 0] = f2bf(w.x);
    t[r][c4 + 1] = f2bf(w.y);
    t[r][c4 + 2] = f2bf(w.z);
    t[r][c4 + 3] = f2bf(w.w);
    __syncthreads();
    unsigned short* o = &WT[(size_t)(nb + r) * 1024 + kb + c4];
    o[0] = t[c4 + 0][r];
    o[1] = t[c4 + 1][r];
    o[2] = t[c4 + 2][r];
    o[3] = t[c4 + 3][r];
}

// ---------------------------------------------------------------------------
// bf16 WMMA GEMM, no LDS: C[M,1024] = epi(A[M,1024] @ WT^T + bias)
//   A  : bf16 row-major [M,1024]
//   WT : bf16 pre-transposed weights [n][k] (L2-resident, B frags contiguous)
// EPI=1: elu(x)+1 ; OUTBF=1: store bf16, else fp32.
// Block = 128 threads (4 waves); wave tile 32x64 (8 accumulators);
// block tile 128x64. 8 WMMAs per 12x16B loads per K-step + A-stream prefetch.
// ---------------------------------------------------------------------------
template<int EPI, int OUTBF>
__global__ __launch_bounds__(128) void gemm_bf16_kernel(
    const unsigned short* __restrict__ A,
    const unsigned short* __restrict__ WT,
    const float* __restrict__ bias,
    void* __restrict__ Cout)
{
    const int tid  = threadIdx.x;
    const int wave = tid >> 5;
    const int lane = tid & 31;
    const int l16  = lane & 15;
    const int half = lane >> 4;

    const int mBase = blockIdx.x * 128 + wave * 32;
    const int nBase = blockIdx.y * 64;

    const unsigned short* A0 = A  + (size_t)(mBase + l16) * 1024;
    const unsigned short* A1 = A0 + (size_t)16 * 1024;
    const unsigned short* Wp[4];
    #pragma unroll
    for (int j = 0; j < 4; ++j)
        Wp[j] = WT + (size_t)(nBase + 16 * j + l16) * 1024;

    const v8f z8 = {0.f,0.f,0.f,0.f,0.f,0.f,0.f,0.f};
    v8f c0[4] = {z8, z8, z8, z8};    // rows mBase..+15   x 4 col tiles
    v8f c1[4] = {z8, z8, z8, z8};    // rows mBase+16..+31

    for (int kk = 0; kk < 1024; kk += 32) {
        // prefetch the streaming A operand ~4 K-steps ahead (speculative,
        // OOB at the tail is dropped by hardware) -> global_prefetch_b8
        __builtin_prefetch((const void*)(A0 + kk + 128), 0, 3);
        __builtin_prefetch((const void*)(A1 + kk + 128), 0, 3);

        v16bf a0 = ldA_g(A0, kk, half);
        v16bf a1 = ldA_g(A1, kk, half);
        #pragma unroll
        for (int j = 0; j < 4; ++j) {
            v16bf b = ldB_g(Wp[j], kk, half);
            c0[j] = WMMA_BF16(a0, b, c0[j]);
            c1[j] = WMMA_BF16(a1, b, c1[j]);
        }
    }

    float bs[4];
    #pragma unroll
    for (int j = 0; j < 4; ++j) bs[j] = bias[nBase + 16 * j + l16];

    #pragma unroll
    for (int r = 0; r < 8; ++r) {
        const int row0 = mBase + r + 8 * half;   // D layout: VGPR r -> M=r (+8 lanes 16..31)
        const int row1 = row0 + 16;
        #pragma unroll
        for (int j = 0; j < 4; ++j) {
            const int col = nBase + 16 * j + l16;
            float v0 = c0[j][r] + bs[j];
            float v1 = c1[j][r] + bs[j];
            if (EPI == 1) {
                v0 = (v0 > 0.f) ? v0 + 1.f : __expf(v0);
                v1 = (v1 > 0.f) ? v1 + 1.f : __expf(v1);
            }
            if (OUTBF) {
                unsigned short* C = (unsigned short*)Cout;
                C[(size_t)row0 * 1024 + col] = f2bf(v0);
                C[(size_t)row1 * 1024 + col] = f2bf(v1);
            } else {
                float* C = (float*)Cout;
                C[(size_t)row0 * 1024 + col] = v0;
                C[(size_t)row1 * 1024 + col] = v1;
            }
        }
    }
}

// ---------------------------------------------------------------------------
// Chunked causal linear attention (chunk C = 64), all matrix work in WMMA.
// Per (b,h) block, 256 threads = 8 waves. Per chunk:
//   G    = Qc @ Vc^T              (then causal mask s<=t -> Gm, bf16 in LDS)
//   num  = Qc @ Sprev^T + Gm @ Kc
//   S   += Kc^T @ Vc
//   z_t  = Zprev + cumsum_{s<=t} Kc[s,j]
//   out[t,j] = num[t,j] / (q_t[j]*z_t[j] + eps)
// Reference quirks preserved: S indexed [j][d] (label swap), inclusive mask,
// elementwise denominator on featured q.
// ---------------------------------------------------------------------------
__global__ __launch_bounds__(256) void chunked_attn_kernel(
    const unsigned short* __restrict__ Qp,
    const unsigned short* __restrict__ Kp,
    const unsigned short* __restrict__ Vp,
    unsigned short* __restrict__ Vlin)
{
    const int bh = blockIdx.x;            // 0..63
    const int b  = bh >> 4;
    const int h  = bh & 15;

    const int tid  = threadIdx.x;
    const int wave = tid >> 5;
    const int lane = tid & 31;
    const int l16  = lane & 15;
    const int half = lane >> 4;
    const int band = wave >> 1;           // 0..3  : 16-row output band
    const int nh   = wave & 1;            // 0..1  : 32-col half

    __shared__ unsigned short Qc [64 * 64];   // [t][d]
    __shared__ unsigned short Vc [64 * 64];   // [s][d]
    __shared__ unsigned short VcT[64 * 64];   // [d][s]
    __shared__ unsigned short KcT[64 * 64];   // [j][s]
    __shared__ unsigned short Gm [64 * 64];   // [t][s] masked
    __shared__ unsigned short Ss [64 * 64];   // state [j][d] (bf16)
    __shared__ unsigned short Cz [64 * 64];   // cumZ [t][j]  (bf16)
    __shared__ float Zp[64];

    for (int i = tid; i < 64 * 64; i += 256) Ss[i] = 0;
    if (tid < 64) Zp[tid] = 0.f;

    const int sr  = tid >> 2;             // staging row 0..63
    const int seg = tid & 3;              // 16-col segment
    const size_t headCol = (size_t)h * HEAD_D;

    union U16 { uint4 u; unsigned short s[8]; };
    const v8f z8 = {0.f,0.f,0.f,0.f,0.f,0.f,0.f,0.f};

    for (int c = 0; c < N_CHUNK; ++c) {
        __syncthreads();   // protect LDS reuse (also makes Ss init visible on c==0)

        // ---- stage chunk tiles; prefetch next chunk's rows for the L0 ----
        {
            const size_t grow = ((size_t)b * SEQ_L + (size_t)c * 64 + sr) * D_MODEL
                                + headCol + (size_t)seg * 16;
            U16 q0, q1, v0, v1, k0, k1;
            q0.u = *reinterpret_cast<const uint4*>(Qp + grow);
            q1.u = *reinterpret_cast<const uint4*>(Qp + grow + 8);
            v0.u = *reinterpret_cast<const uint4*>(Vp + grow);
            v1.u = *reinterpret_cast<const uint4*>(Vp + grow + 8);
            k0.u = *reinterpret_cast<const uint4*>(Kp + grow);
            k1.u = *reinterpret_cast<const uint4*>(Kp + grow + 8);
            if (c + 1 < N_CHUNK) {
                const size_t gn = grow + (size_t)64 * D_MODEL;
                __builtin_prefetch((const void*)(Qp + gn), 0, 3);
                __builtin_prefetch((const void*)(Kp + gn), 0, 3);
                __builtin_prefetch((const void*)(Vp + gn), 0, 3);
            }
            *reinterpret_cast<uint4*>(&Qc[sr * 64 + seg * 16])     = q0.u;
            *reinterpret_cast<uint4*>(&Qc[sr * 64 + seg * 16 + 8]) = q1.u;
            *reinterpret_cast<uint4*>(&Vc[sr * 64 + seg * 16])     = v0.u;
            *reinterpret_cast<uint4*>(&Vc[sr * 64 + seg * 16 + 8]) = v1.u;
            #pragma unroll
            for (int i = 0; i < 8; ++i) {
                const int cc0 = seg * 16 + i, cc1 = seg * 16 + 8 + i;
                VcT[cc0 * 64 + sr] = v0.s[i];
                VcT[cc1 * 64 + sr] = v1.s[i];
                KcT[cc0 * 64 + sr] = k0.s[i];
                KcT[cc1 * 64 + sr] = k1.s[i];
            }
        }
        __syncthreads();   // tiles ready; Ss holds S_prev

        // ---- column cumsum of K: Cz[t][j] = Zp[j] + sum_{s<=t} K[s,j] ----
        if (tid < 64) {
            float run = Zp[tid];
            const unsigned short* kr = &KcT[tid * 64];
            #pragma unroll 4
            for (int t = 0; t < 64; ++t) {
                run += bf2f(kr[t]);
                Cz[t * 64 + tid] = f2bf(run);
            }
            Zp[tid] = run;
        }

        // ---- G = Qc @ Vc^T  and  num1 = Qc @ Sprev^T ----
        v8f g0 = z8, g1 = z8, n0 = z8, n1 = z8;
        #pragma unroll
        for (int kt = 0; kt < 2; ++kt) {
            v16bf a   = ldA64(Qc, band * 16 + l16, kt, half);
            v16bf bv0 = ldB64(Vc, nh * 32 + l16,      kt, half);
            v16bf bv1 = ldB64(Vc, nh * 32 + 16 + l16, kt, half);
            v16bf bs0 = ldB64(Ss, nh * 32 + l16,      kt, half);
            v16bf bs1 = ldB64(Ss, nh * 32 + 16 + l16, kt, half);
            g0 = WMMA_BF16(a, bv0, g0);
            g1 = WMMA_BF16(a, bv1, g1);
            n0 = WMMA_BF16(a, bs0, n0);
            n1 = WMMA_BF16(a, bs1, n1);
        }
        // causal mask (inclusive) and stash G as bf16
        #pragma unroll
        for (int r = 0; r < 8; ++r) {
            const int t  = band * 16 + r + 8 * half;
            const int s0 = nh * 32 + l16;
            const int s1 = s0 + 16;
            Gm[t * 64 + s0] = (s0 <= t) ? f2bf(g0[r]) : (unsigned short)0;
            Gm[t * 64 + s1] = (s1 <= t) ? f2bf(g1[r]) : (unsigned short)0;
        }
        __syncthreads();   // Gm, Cz ready; every wave finished reading S_prev

        // ---- num2 += Gm @ Kc  and  Supd = Kc^T @ Vc ----
        v8f u0 = z8, u1 = z8;
        #pragma unroll
        for (int kt = 0; kt < 2; ++kt) {
            v16bf ag  = ldA64(Gm, band * 16 + l16, kt, half);
            v16bf bk0 = ldB64(KcT, nh * 32 + l16,      kt, half);
            v16bf bk1 = ldB64(KcT, nh * 32 + 16 + l16, kt, half);
            n0 = WMMA_BF16(ag, bk0, n0);
            n1 = WMMA_BF16(ag, bk1, n1);
            v16bf ak  = ldA64(KcT, band * 16 + l16, kt, half);
            v16bf bt0 = ldB64(VcT, nh * 32 + l16,      kt, half);
            v16bf bt1 = ldB64(VcT, nh * 32 + 16 + l16, kt, half);
            u0 = WMMA_BF16(ak, bt0, u0);
            u1 = WMMA_BF16(ak, bt1, u1);
        }

        // ---- S += Kc^T @ Vc  (each element owned by exactly one lane) ----
        #pragma unroll
        for (int r = 0; r < 8; ++r) {
            const int j  = band * 16 + r + 8 * half;
            const int d0 = nh * 32 + l16;
            const int d1 = d0 + 16;
            Ss[j * 64 + d0] = f2bf(bf2f(Ss[j * 64 + d0]) + u0[r]);
            Ss[j * 64 + d1] = f2bf(bf2f(Ss[j * 64 + d1]) + u1[r]);
        }

        // ---- epilogue: out[t,j] = num / (q*z + eps), bf16 store ----
        #pragma unroll
        for (int r = 0; r < 8; ++r) {
            const int t  = band * 16 + r + 8 * half;
            const int j0 = nh * 32 + l16;
            const int j1 = j0 + 16;
            const size_t orow = ((size_t)b * SEQ_L + (size_t)c * 64 + t) * D_MODEL + headCol;
            const float q0f = bf2f(Qc[t * 64 + j0]);
            const float q1f = bf2f(Qc[t * 64 + j1]);
            const float zz0 = bf2f(Cz[t * 64 + j0]);
            const float zz1 = bf2f(Cz[t * 64 + j1]);
            Vlin[orow + j0] = f2bf(n0[r] / (q0f * zz0 + EPS_CAUSAL));
            Vlin[orow + j1] = f2bf(n1[r] / (q1f * zz1 + EPS_CAUSAL));
        }
    }
}

// ---------------------------------------------------------------------------
extern "C" void kernel_launch(void* const* d_in, const int* in_sizes, int n_in,
                              void* d_out, int out_size, void* d_ws, size_t ws_size,
                              hipStream_t stream) {
    const float* queries = (const float*)d_in[0];
    const float* keys    = (const float*)d_in[1];
    const float* values  = (const float*)d_in[2];
    const float* Wq      = (const float*)d_in[3];
    const float* bq      = (const float*)d_in[4];
    const float* Wk      = (const float*)d_in[5];
    const float* bk      = (const float*)d_in[6];
    const float* Wv      = (const float*)d_in[7];
    const float* bv      = (const float*)d_in[8];
    const float* Wo      = (const float*)d_in[9];
    const float* bo      = (const float*)d_in[10];
    float* out = (float*)d_out;

    const size_t plane = (size_t)M_ROWS * D_MODEL;   // 8M elems
    const size_t wmat  = (size_t)D_MODEL * D_MODEL;  // 1M elems
    unsigned short* ws16 = (unsigned short*)d_ws;

    unsigned short* Xq   = ws16;                     // activations, bf16
    unsigned short* Xk   = Xq + plane;
    unsigned short* Xv   = Xk + plane;
    unsigned short* WqT  = Xv + plane;               // transposed weights, bf16
    unsigned short* WkT  = WqT + wmat;
    unsigned short* WvT  = WkT + wmat;
    unsigned short* WoT  = WvT + wmat;
    unsigned short* Qp   = WoT + wmat;               // featured projections, bf16
    unsigned short* Kp   = Qp + plane;
    unsigned short* Vp   = Kp + plane;
    unsigned short* Vlin = Vp + plane;               // attention output, bf16
    // total: (7*8M + 4*1M) * 2 B = 120 MB of d_ws

    // --- prep: bf16 conversions + weight transposes ---
    {
        dim3 g((unsigned)(plane / (256 * 4))), blk(256);
        cvt_bf16_kernel<<<g, blk, 0, stream>>>(queries, Xq, (int)plane);
        cvt_bf16_kernel<<<g, blk, 0, stream>>>(keys,    Xk, (int)plane);
        cvt_bf16_kernel<<<g, blk, 0, stream>>>(values,  Xv, (int)plane);
        dim3 gt(32, 32), bt(256);
        transpose_w_kernel<<<gt, bt, 0, stream>>>(Wq, WqT);
        transpose_w_kernel<<<gt, bt, 0, stream>>>(Wk, WkT);
        transpose_w_kernel<<<gt, bt, 0, stream>>>(Wv, WvT);
        transpose_w_kernel<<<gt, bt, 0, stream>>>(Wo, WoT);
    }

    // --- projections (feat on Q,K), bf16 out ---
    dim3 grid(M_ROWS / 128, D_MODEL / 64), block(128);
    gemm_bf16_kernel<1, 1><<<grid, block, 0, stream>>>(Xq, WqT, bq, Qp);
    gemm_bf16_kernel<1, 1><<<grid, block, 0, stream>>>(Xk, WkT, bk, Kp);
    gemm_bf16_kernel<0, 1><<<grid, block, 0, stream>>>(Xv, WvT, bv, Vp);

    // --- chunked causal linear attention (WMMA) ---
    chunked_attn_kernel<<<dim3(BATCH * N_HEADS), dim3(256), 0, stream>>>(Qp, Kp, Vp, Vlin);

    // --- output projection, fp32 out ---
    gemm_bf16_kernel<0, 0><<<grid, block, 0, stream>>>(Vlin, WoT, bo, out);
}
